// CoarseMatching_29798483099798
// MI455X (gfx1250) — matrix operand order repaired and checked by reading, not compiled, not run on hardware
//
#include <hip/hip_runtime.h>

typedef __bf16 bf16_t;
typedef __attribute__((ext_vector_type(16))) __bf16 v16bf;
typedef __attribute__((ext_vector_type(8)))  float  v8f;
typedef __attribute__((ext_vector_type(4)))  float  v4f;

#define BATCH    4
#define L0DIM    2048
#define L1DIM    1024
#define CDIM     64
#define GRIDN    12
#define G3       1728
#define VOXEL    0.12f
#define OFFS     0.72f
#define TM       16                    // l0 rows per workgroup (one WMMA M-tile)
#define WAVES    8
#define NTHREADS 256
#define COLS_PER_WAVE  (L1DIM / WAVES)        // 128
#define TILES_PER_WAVE (COLS_PER_WAVE / 16)   // 8

__global__ __launch_bounds__(NTHREADS) void coarse_matching_fused(
    const float* __restrict__ x0, const float* __restrict__ x1,
    const float* __restrict__ pos0, const float* __restrict__ pos1,
    float* __restrict__ out_flow, float* __restrict__ out_grid)
{
  // ---- LDS: 16 private 12^3 grids + A-tile + reduction scratch (~113 KB of 320 KB) ----
  __shared__ float lds_grid[TM * G3];                       // 110592 B
  __shared__ __align__(32) bf16_t lds_a[2][32][16];         // A frags, WMMA lane layout
  __shared__ float lds_red[WAVES][TM];
  __shared__ float lds_rowmax[TM];
  __shared__ float lds_rowinv[TM];
  __shared__ float lds_flow[TM][3];
  __shared__ float lds_pos0[TM][3];

  const int tid  = threadIdx.x;
  const int wave = tid >> 5;
  const int lane = tid & 31;
  const int h    = lane >> 4;      // lane half
  const int ln   = lane & 15;

  const int wgs_per_b = L0DIM / TM;            // 128
  const int b  = blockIdx.x / wgs_per_b;
  const int m0 = (blockIdx.x % wgs_per_b) * TM;

  // ---- init LDS ----
  for (int i = tid; i < TM * G3; i += NTHREADS) lds_grid[i] = 0.0f;
  if (tid < TM * 3) {
    lds_flow[tid / 3][tid % 3] = 0.0f;
    lds_pos0[tid / 3][tid % 3] = pos0[((size_t)b * L0DIM + m0 + tid / 3) * 3 + tid % 3];
  }
  // ---- stage A tile (x0, f32 -> bf16) directly in WMMA A-fragment lane layout ----
  // lds_a[s][l][e] = x0[b][m0 + (l&15)][32*s + e + ((l>>4)<<4)]
  for (int i = tid; i < 2 * 32 * 16; i += NTHREADS) {
    const int e  = i & 15;
    const int l  = (i >> 4) & 31;
    const int s  = i >> 9;
    const int m  = l & 15;
    const int hh = l >> 4;
    const int k  = 32 * s + e + (hh << 4);
    lds_a[s][l][e] = (bf16_t)x0[((size_t)b * L0DIM + m0 + m) * CDIM + k];
  }
  __syncthreads();

  const v16bf a0 = *(const v16bf*)(&lds_a[0][lane][0]);
  const v16bf a1 = *(const v16bf*)(&lds_a[1][lane][0]);

  // ---- GEMM: 16 x 1024 scores, kept resident in accumulator VGPRs ----
  v8f acc[TILES_PER_WAVE];
  const int nbase_w = wave * COLS_PER_WAVE;
#pragma unroll
  for (int t = 0; t < TILES_PER_WAVE; ++t) {
    const int n = nbase_w + t * 16 + ln;                 // B-frag column N = lane&15
    const float* xr = &x1[((size_t)b * L1DIM + n) * CDIM + h * 16];
    // B-frag element e: K = 32*s + e + 16*h  -> 16 consecutive f32 per lane per k-step
    v4f q0 = *(const v4f*)(xr + 0);
    v4f q1 = *(const v4f*)(xr + 4);
    v4f q2 = *(const v4f*)(xr + 8);
    v4f q3 = *(const v4f*)(xr + 12);
    v4f r0 = *(const v4f*)(xr + 32);
    v4f r1 = *(const v4f*)(xr + 36);
    v4f r2 = *(const v4f*)(xr + 40);
    v4f r3 = *(const v4f*)(xr + 44);
    v16bf bm0, bm1;
#pragma unroll
    for (int e = 0; e < 4; ++e) {
      bm0[e]      = (bf16_t)q0[e];
      bm0[e + 4]  = (bf16_t)q1[e];
      bm0[e + 8]  = (bf16_t)q2[e];
      bm0[e + 12] = (bf16_t)q3[e];
      bm1[e]      = (bf16_t)r0[e];
      bm1[e + 4]  = (bf16_t)r1[e];
      bm1[e + 8]  = (bf16_t)r2[e];
      bm1[e + 12] = (bf16_t)r3[e];
    }
    v8f c = {0.f, 0.f, 0.f, 0.f, 0.f, 0.f, 0.f, 0.f};
    c = __builtin_amdgcn_wmma_f32_16x16x32_bf16(false, a0, false, bm0, (short)0, c, false, false);
    c = __builtin_amdgcn_wmma_f32_16x16x32_bf16(false, a1, false, bm1, (short)0, c, false, false);
#pragma unroll
    for (int r = 0; r < 8; ++r) c[r] *= 0.125f;          // 1/sqrt(64)
    acc[t] = c;
  }

  // ---- softmax: row max (xor-shuffle within 16-lane half, then cross-wave via LDS) ----
#pragma unroll
  for (int r = 0; r < 8; ++r) {
    float v = acc[0][r];
#pragma unroll
    for (int t = 1; t < TILES_PER_WAVE; ++t) v = fmaxf(v, acc[t][r]);
    v = fmaxf(v, __shfl_xor(v, 1, 32));
    v = fmaxf(v, __shfl_xor(v, 2, 32));
    v = fmaxf(v, __shfl_xor(v, 4, 32));
    v = fmaxf(v, __shfl_xor(v, 8, 32));
    if (ln == 0) lds_red[wave][r + 8 * h] = v;
  }
  __syncthreads();
  if (tid < TM) {
    float v = lds_red[0][tid];
    for (int w = 1; w < WAVES; ++w) v = fmaxf(v, lds_red[w][tid]);
    lds_rowmax[tid] = v;
  }
  __syncthreads();

  // ---- exp in-place + row sum ----
#pragma unroll
  for (int r = 0; r < 8; ++r) {
    const float rm = lds_rowmax[r + 8 * h];
    float s = 0.0f;
#pragma unroll
    for (int t = 0; t < TILES_PER_WAVE; ++t) {
      const float e = __expf(acc[t][r] - rm);
      acc[t][r] = e;                                     // unnormalized prob
      s += e;
    }
    s += __shfl_xor(s, 1, 32);
    s += __shfl_xor(s, 2, 32);
    s += __shfl_xor(s, 4, 32);
    s += __shfl_xor(s, 8, 32);
    if (ln == 0) lds_red[wave][r + 8 * h] = s;
  }
  __syncthreads();
  if (tid < TM) {
    float s = 0.0f;
    for (int w = 0; w < WAVES; ++w) s += lds_red[w][tid];
    lds_rowinv[tid] = 1.0f / s;
  }
  __syncthreads();

  // ---- fused: correspondence accumulation + trilinear splat into LDS grids ----
  float facc[8][3];
#pragma unroll
  for (int r = 0; r < 8; ++r) { facc[r][0] = 0.f; facc[r][1] = 0.f; facc[r][2] = 0.f; }

  for (int t = 0; t < TILES_PER_WAVE; ++t) {
    const int n = nbase_w + t * 16 + ln;
    const float* p1 = &pos1[((size_t)b * L1DIM + n) * 3];
    const float p1x = p1[0], p1y = p1[1], p1z = p1[2];
#pragma unroll
    for (int r = 0; r < 8; ++r) {
      const int m = r + 8 * h;
      const float p = acc[t][r] * lds_rowinv[m];
      facc[r][0] += p * p1x;
      facc[r][1] += p * p1y;
      facc[r][2] += p * p1z;
      const float cx = (p1x - lds_pos0[m][0] + OFFS) * (1.0f / VOXEL) - 0.5f;
      const float cy = (p1y - lds_pos0[m][1] + OFFS) * (1.0f / VOXEL) - 0.5f;
      const float cz = (p1z - lds_pos0[m][2] + OFFS) * (1.0f / VOXEL) - 0.5f;
      const float bx = floorf(cx), by = floorf(cy), bz = floorf(cz);
      const float fx = cx - bx, fy = cy - by, fz = cz - bz;
      const int ibx = (int)bx, iby = (int)by, ibz = (int)bz;
      float* gm = &lds_grid[m * G3];
#pragma unroll
      for (int cor = 0; cor < 8; ++cor) {
        const int dx = (cor >> 2) & 1, dy = (cor >> 1) & 1, dz = cor & 1;
        const int ix = ibx + dx, iy = iby + dy, iz = ibz + dz;
        const float w = p * (dx ? fx : 1.0f - fx) * (dy ? fy : 1.0f - fy) * (dz ? fz : 1.0f - fz);
        if ((unsigned)ix < (unsigned)GRIDN && (unsigned)iy < (unsigned)GRIDN &&
            (unsigned)iz < (unsigned)GRIDN)
          atomicAdd(&gm[ix * (GRIDN * GRIDN) + iy * GRIDN + iz], w);   // ds_add_f32
      }
    }
  }

  // ---- reduce correspondence across lanes/waves ----
#pragma unroll
  for (int r = 0; r < 8; ++r) {
    const int m = r + 8 * h;
#pragma unroll
    for (int d = 0; d < 3; ++d) {
      float v = facc[r][d];
      v += __shfl_xor(v, 1, 32);
      v += __shfl_xor(v, 2, 32);
      v += __shfl_xor(v, 4, 32);
      v += __shfl_xor(v, 8, 32);
      if (ln == 0) atomicAdd(&lds_flow[m][d], v);
    }
  }
  __syncthreads();

  // ---- outputs ----
  if (tid < TM * 3) {
    const int m = tid / 3, d = tid % 3;
    out_flow[((size_t)b * L0DIM + m0 + m) * 3 + d] = lds_flow[m][d] - lds_pos0[m][d];
  }
  const size_t gbase = ((size_t)b * L0DIM + m0) * G3;
  for (int i = tid; i < TM * G3; i += NTHREADS)
    out_grid[gbase + i] = lds_grid[i];
}

extern "C" void kernel_launch(void* const* d_in, const int* in_sizes, int n_in,
                              void* d_out, int out_size, void* d_ws, size_t ws_size,
                              hipStream_t stream) {
  (void)in_sizes; (void)n_in; (void)out_size; (void)d_ws; (void)ws_size;
  const float* x0   = (const float*)d_in[0];
  const float* x1   = (const float*)d_in[1];
  const float* pos0 = (const float*)d_in[2];
  const float* pos1 = (const float*)d_in[3];
  float* out_flow = (float*)d_out;                                   // (B, L0, 3)
  float* out_grid = out_flow + (size_t)BATCH * L0DIM * 3;            // (B, L0, G3)
  const int nblocks = BATCH * (L0DIM / TM);                          // 512
  coarse_matching_fused<<<nblocks, NTHREADS, 0, stream>>>(x0, x1, pos0, pos1,
                                                          out_flow, out_grid);
}
